// MultitaskRNN_7413113552925
// MI455X (gfx1250) — matrix operand order repaired and checked by reading, not compiled, not run on hardware
//
#include <hip/hip_runtime.h>
#include <hip/hip_bf16.h>
#include <math.h>
#include <stdint.h>

// ---------------------------------------------------------------------------
// MultitaskRNN on MI455X (gfx1250, wave32, WMMA)
//   B=256, T=512, I=128, H=1024, O=64
//   h' = 0.8*h + 0.2*tanh(x_t W_in^T + h W_rec^T + b + noise)
//   y  = h' W_out^T + b_out
// bf16 WMMA (v_wmma_f32_16x16x32_bf16), fp32 accumulate, fp32 state.
// LDS-staged 64x64 macro-tiles (double buffered, async-to-LDS when available).
// ---------------------------------------------------------------------------

typedef __attribute__((ext_vector_type(16))) __bf16 v16bf;
typedef __attribute__((ext_vector_type(8)))  __bf16 v8bf;
typedef __attribute__((ext_vector_type(8)))  float  v8f;
typedef __attribute__((ext_vector_type(4)))  int    v4i;

constexpr int BB = 256;
constexpr int TT = 512;
constexpr int II = 128;
constexpr int HH = 1024;
constexpr int OO = 64;

constexpr int LDS_STRIDE = 40;   // elements per staged row (32 + 8 pad -> 80B)

#define ALPHA_F    0.2f
#define OMA_F      0.8f
#define SIGMA_F    0.15811388300841898f   /* sqrt(2/alpha) * 0.05 */

#ifndef __has_builtin
#define __has_builtin(x) 0
#endif

#if __has_builtin(__builtin_amdgcn_global_load_async_to_lds_b128)
#define ASYNC_LDS 1
#else
#define ASYNC_LDS 0
#endif

#if ASYNC_LDS
typedef __attribute__((address_space(1))) v4i* gp4;   // global int4*
typedef __attribute__((address_space(3))) v4i* lp4;   // LDS int4*
__device__ __forceinline__ gp4 to_global4(const void* p) {
    return (gp4)(unsigned long long)(uintptr_t)p;
}
__device__ __forceinline__ lp4 to_lds4(const void* p) {
    // generic LDS address: low 32 bits are the LDS offset (ISA 10.2 aperture rule)
    return (lp4)(unsigned int)(uintptr_t)p;
}
#endif

__device__ __forceinline__ void wait_async_then_barrier() {
#if ASYNC_LDS
#if __has_builtin(__builtin_amdgcn_s_wait_asynccnt)
    __builtin_amdgcn_s_wait_asynccnt(0);
#else
    asm volatile("s_wait_asynccnt 0x0" ::: "memory");
#endif
#endif
    __syncthreads();
}

// ---- counter-based Gaussian noise (splitmix64 + Box-Muller) ---------------
__device__ __forceinline__ float rng_normal(unsigned int idx) {
    unsigned long long z = (unsigned long long)idx * 0x9E3779B97F4A7C15ull
                         + 0xD1B54A32D192ED03ull;
    z ^= z >> 30; z *= 0xBF58476D1CE4E5B9ull;
    z ^= z >> 27; z *= 0x94D049BB133111EBull;
    z ^= z >> 31;
    unsigned int u1 = (unsigned int)z;
    unsigned int u2 = (unsigned int)(z >> 32);
    float f1 = (float)(u1 >> 8) * (1.0f / 16777216.0f) + (0.5f / 16777216.0f);
    float f2 = (float)(u2 >> 8) * (1.0f / 16777216.0f);
    float r  = sqrtf(-2.0f * logf(f1));
    return r * __cosf(6.28318530717958647f * f2);
}

// ---- WMMA fragment loaders (wave32, contiguous k = 16*(L/16) + s for A and B) --------------
// A 16x32: lane m=L&15; L<16 holds K 0-15,        L>=16 holds K 16-31
// B 32x16: lane n=L&15; L<16 holds K 0-15,        L>=16 holds K 16-31
__device__ __forceinline__ v16bf frag_a_lds(const __bf16* tile, int lane) {
    const int row = lane & 15;
    const int kb  = (lane & 16) ? 16 : 0;
    const __bf16* p = tile + row * LDS_STRIDE + kb;
    v8bf lo = *(const v8bf*)(p);
    v8bf hi = *(const v8bf*)(p + 8);
    v16bf a;
#pragma unroll
    for (int j = 0; j < 8; ++j) { a[j] = lo[j]; a[8 + j] = hi[j]; }
    return a;
}

__device__ __forceinline__ v16bf frag_b_lds(const __bf16* tile, int lane) {
    const int row = lane & 15;
    const int kb  = (lane & 16) ? 16 : 0;
    const __bf16* p = tile + row * LDS_STRIDE + kb;
    v8bf lo = *(const v8bf*)(p);
    v8bf hi = *(const v8bf*)(p + 8);
    v16bf b;
#pragma unroll
    for (int j = 0; j < 8; ++j) { b[j] = lo[j]; b[8 + j] = hi[j]; }
    return b;
}

// ---- stage one 64-row x 32-col bf16 chunk pair into LDS -------------------
// 256 threads: each moves one 16B beat of A and one of B.
__device__ __forceinline__ void stage_chunk(__bf16* dstA, __bf16* dstB,
                                            const __bf16* __restrict__ srcA,
                                            const __bf16* __restrict__ srcB,
                                            int tid) {
    const int r  = tid >> 2;          // 0..63  (row)
    const int cb = (tid & 3) * 8;     // 0,8,16,24 (element offset)
    const __bf16* ga = srcA + r * HH + cb;
    const __bf16* gb = srcB + r * HH + cb;
    __bf16* la = dstA + r * LDS_STRIDE + cb;
    __bf16* lb = dstB + r * LDS_STRIDE + cb;
#if ASYNC_LDS
    __builtin_amdgcn_global_load_async_to_lds_b128(to_global4(ga), to_lds4(la), 0, 0);
    __builtin_amdgcn_global_load_async_to_lds_b128(to_global4(gb), to_lds4(lb), 0, 0);
#else
    *(v8bf*)la = *(const v8bf*)ga;
    *(v8bf*)lb = *(const v8bf*)gb;
#endif
}

// ---- output projection tile: out[b, t, :] = h @ W_out^T + b_out -----------
__device__ __forceinline__ void out_tile(int w, int lane,
                                         const __bf16* __restrict__ h_b,
                                         const __bf16* __restrict__ wout_b,
                                         const float* __restrict__ b_out,
                                         float* __restrict__ out, int t) {
    const int m0 = (w >> 2) * 16;                // batch tile
    const int n0 = (w & 3) * 16;                 // output tile
    const int akb = (lane & 16) ? 16 : 0;
    const int bkb = (lane & 16) ? 16 : 0;
    const __bf16* aRow = h_b    + (m0 + (lane & 15)) * HH + akb;
    const __bf16* bRow = wout_b + (n0 + (lane & 15)) * HH + bkb;

    v8f c = {0.f, 0.f, 0.f, 0.f, 0.f, 0.f, 0.f, 0.f};
    for (int k0 = 0; k0 < HH; k0 += 32) {
        v8bf lo = *(const v8bf*)(aRow + k0);
        v8bf hi = *(const v8bf*)(aRow + k0 + 8);
        v16bf a;
#pragma unroll
        for (int j = 0; j < 8; ++j) { a[j] = lo[j]; a[8 + j] = hi[j]; }
        v16bf bb = *(const v16bf*)(bRow + k0);     // 32B aligned
        c = __builtin_amdgcn_wmma_f32_16x16x32_bf16(false, a, false, bb,
                                                    (short)0, c, false, false);
    }
    const int gn  = n0 + (lane & 15);
    const int mhi = (lane & 16) ? 8 : 0;
    const float bo = b_out[gn];
#pragma unroll
    for (int v = 0; v < 8; ++v) {
        const int gm = m0 + mhi + v;
        out[((size_t)gm * TT + t) * OO + gn] = c[v] + bo;
    }
}

// ---------------------------------------------------------------------------
// Fused per-timestep kernel, grid = 72 blocks x 256 threads:
//  blocks [0,64):  h_{t+1} from h_t   (64x64 macro-tile / block, LDS staged)
//  blocks [64,72): out[t-1] from h_t  (skipped when t==0)
// ---------------------------------------------------------------------------
__global__ void __launch_bounds__(256)
rnn_fused_kernel(const float* __restrict__ x,        // [B,T,I] fp32
                 int t,
                 const float* __restrict__ h_in_f,   // [B,H] fp32
                 const __bf16* __restrict__ h_in_b,  // [B,H] bf16
                 const __bf16* __restrict__ wrec_b,  // [H,H] bf16
                 const __bf16* __restrict__ win_b,   // [H,I] bf16
                 const float* __restrict__ bias,     // [H]
                 float* __restrict__ h_out_f,        // [B,H] fp32
                 __bf16* __restrict__ h_out_b,       // [B,H] bf16
                 const __bf16* __restrict__ wout_b,  // [O,H] bf16
                 const float* __restrict__ b_out,    // [O]
                 float* __restrict__ out)            // [B,T,O]
{
    __shared__ __bf16 ldsA[2][64 * LDS_STRIDE];     // 64 x 32 chunk of h
    __shared__ __bf16 ldsB[2][64 * LDS_STRIDE];     // 64 x 32 chunk of W_rec

    const int tid  = threadIdx.x;
    const int lane = tid & 31;
    const int wave = tid >> 5;

    if (blockIdx.x >= 64) {
        if (t == 0) return;
        out_tile((blockIdx.x - 64) * 8 + wave, lane, h_in_b, wout_b, b_out, out, t - 1);
        return;
    }

    const int mg = blockIdx.x >> 4;     // 0..3  -> m_base
    const int ng = blockIdx.x & 15;     // 0..15 -> n_base
    const int m_base = mg * 64;
    const int n_base = ng * 64;
    const int p = wave >> 2;            // 0..1  m-pair within block
    const int q = wave & 3;             // 0..3  n-tile within block

    const __bf16* srcA = h_in_b + (size_t)m_base * HH;
    const __bf16* srcB = wrec_b + (size_t)n_base * HH;

    // ---- recurrent GEMM over K = 1024, LDS double buffered ----
    stage_chunk(ldsA[0], ldsB[0], srcA, srcB, tid);
    wait_async_then_barrier();

    v8f c0 = {0.f, 0.f, 0.f, 0.f, 0.f, 0.f, 0.f, 0.f};
    v8f c1 = {0.f, 0.f, 0.f, 0.f, 0.f, 0.f, 0.f, 0.f};

    for (int kc = 0; kc < 32; ++kc) {
        const int buf = kc & 1;
        if (kc + 1 < 32)
            stage_chunk(ldsA[buf ^ 1], ldsB[buf ^ 1],
                        srcA + (kc + 1) * 32, srcB + (kc + 1) * 32, tid);

        const __bf16* tA = &ldsA[buf][(p * 32) * LDS_STRIDE];
        const __bf16* tB = &ldsB[buf][(q * 16) * LDS_STRIDE];
        v16bf bf = frag_b_lds(tB, lane);
        v16bf a0 = frag_a_lds(tA, lane);
        v16bf a1 = frag_a_lds(tA + 16 * LDS_STRIDE, lane);
        c0 = __builtin_amdgcn_wmma_f32_16x16x32_bf16(false, a0, false, bf,
                                                     (short)0, c0, false, false);
        c1 = __builtin_amdgcn_wmma_f32_16x16x32_bf16(false, a1, false, bf,
                                                     (short)0, c1, false, false);
        wait_async_then_barrier();
    }

    // ---- input projection over K = 128 (direct, fp32 -> bf16 on the fly) ----
    const int am  = m_base + p * 32 + (lane & 15);
    const int akb = (lane & 16) ? 16 : 0;
    const int bn  = n_base + q * 16 + (lane & 15);
    const int bkb = (lane & 16) ? 16 : 0;
    const __bf16* binRow = win_b + bn * II + bkb;
    for (int k0 = 0; k0 < II; k0 += 32) {
        v16bf bb = *(const v16bf*)(binRow + k0);   // 32B aligned
#pragma unroll
        for (int j = 0; j < 2; ++j) {
            const float* xR = x + (size_t)(am + j * 16) * (TT * II)
                                + (size_t)t * II + akb + k0;
            v16bf a;
#pragma unroll
            for (int e = 0; e < 8; ++e) {
                a[e]     = (__bf16)xR[e];
                a[8 + e] = (__bf16)xR[8 + e];
            }
            if (j == 0)
                c0 = __builtin_amdgcn_wmma_f32_16x16x32_bf16(false, a, false, bb,
                                                             (short)0, c0, false, false);
            else
                c1 = __builtin_amdgcn_wmma_f32_16x16x32_bf16(false, a, false, bb,
                                                             (short)0, c1, false, false);
        }
    }

    // ---- epilogue: bias + noise + tanh + leaky blend (fp32 state) ----
    const int gn  = n_base + q * 16 + (lane & 15);
    const int mhi = (lane & 16) ? 8 : 0;
    const float bv = bias[gn];
#pragma unroll
    for (int j = 0; j < 2; ++j) {
        const v8f& c = (j == 0) ? c0 : c1;
#pragma unroll
        for (int v = 0; v < 8; ++v) {
            const int gm = m_base + p * 32 + j * 16 + mhi + v;
            const unsigned int idx =
                ((unsigned)t * BB + (unsigned)gm) * HH + (unsigned)gn;
            float pre  = c[v] + bv + SIGMA_F * rng_normal(idx);
            float hold = h_in_f[gm * HH + gn];
            float hn   = OMA_F * hold + ALPHA_F * tanhf(pre);
            h_out_f[gm * HH + gn] = hn;
            h_out_b[gm * HH + gn] = (__bf16)hn;
        }
    }
}

// ---- final output projection (t = T-1) ------------------------------------
__global__ void __launch_bounds__(256)
rnn_out_final(const __bf16* __restrict__ h_b,
              const __bf16* __restrict__ wout_b,
              const float* __restrict__ b_out,
              float* __restrict__ out, int t) {
    out_tile(blockIdx.x * 8 + (threadIdx.x >> 5), threadIdx.x & 31,
             h_b, wout_b, b_out, out, t);
}

// ---- prep kernels ---------------------------------------------------------
__global__ void cvt_f32_bf16(const float* __restrict__ src,
                             __bf16* __restrict__ dst, int n) {
    int i = blockIdx.x * blockDim.x + threadIdx.x;
    if (i < n) dst[i] = (__bf16)src[i];
}

__global__ void init_h_kernel(const float* __restrict__ hidden,
                              float* __restrict__ hf,
                              __bf16* __restrict__ hb, int n) {
    int i = blockIdx.x * blockDim.x + threadIdx.x;
    if (i < n) { float v = hidden[i]; hf[i] = v; hb[i] = (__bf16)v; }
}

// ---------------------------------------------------------------------------
extern "C" void kernel_launch(void* const* d_in, const int* in_sizes, int n_in,
                              void* d_out, int out_size, void* d_ws, size_t ws_size,
                              hipStream_t stream) {
    const float* inputs = (const float*)d_in[0];   // [B,T,I]
    const float* hidden = (const float*)d_in[1];   // [B,H]
    const float* W_in   = (const float*)d_in[2];   // [H,I]
    const float* W_rec  = (const float*)d_in[3];   // [H,H]
    const float* bvec   = (const float*)d_in[4];   // [H]
    const float* W_out  = (const float*)d_in[5];   // [O,H]
    const float* b_out  = (const float*)d_in[6];   // [O]
    float* out = (float*)d_out;                    // [B,T,O] ++ [B,H]

    char* ws = (char*)d_ws;
    size_t off = 0;
    auto carve = [&](size_t bytes) -> void* {
        void* p = ws + off;
        off = (off + bytes + 255) & ~(size_t)255;
        return p;
    };
    __bf16* wrec_b = (__bf16*)carve((size_t)HH * HH * sizeof(__bf16));
    __bf16* win_b  = (__bf16*)carve((size_t)HH * II * sizeof(__bf16));
    __bf16* wout_b = (__bf16*)carve((size_t)OO * HH * sizeof(__bf16));
    float*  hf[2];
    __bf16* hb[2];
    hf[0] = (float*)carve((size_t)BB * HH * sizeof(float));
    hf[1] = (float*)carve((size_t)BB * HH * sizeof(float));
    hb[0] = (__bf16*)carve((size_t)BB * HH * sizeof(__bf16));
    hb[1] = (__bf16*)carve((size_t)BB * HH * sizeof(__bf16));
    (void)ws_size; (void)in_sizes; (void)n_in; (void)out_size;

    cvt_f32_bf16<<<(HH * HH + 255) / 256, 256, 0, stream>>>(W_rec, wrec_b, HH * HH);
    cvt_f32_bf16<<<(HH * II + 255) / 256, 256, 0, stream>>>(W_in,  win_b,  HH * II);
    cvt_f32_bf16<<<(OO * HH + 255) / 256, 256, 0, stream>>>(W_out, wout_b, OO * HH);
    init_h_kernel<<<(BB * HH + 255) / 256, 256, 0, stream>>>(hidden, hf[0], hb[0], BB * HH);

    int cur = 0;
    for (int t = 0; t < TT; ++t) {
        int nxt = cur ^ 1;
        rnn_fused_kernel<<<72, 256, 0, stream>>>(inputs, t,
                                                 hf[cur], hb[cur],
                                                 wrec_b, win_b, bvec,
                                                 hf[nxt], hb[nxt],
                                                 wout_b, b_out, out);
        cur = nxt;
    }
    rnn_out_final<<<8, 256, 0, stream>>>(hb[cur], wout_b, b_out, out, TT - 1);

    (void)hipMemcpyAsync(out + (size_t)BB * TT * OO, hf[cur],
                         (size_t)BB * HH * sizeof(float),
                         hipMemcpyDeviceToDevice, stream);
}
